// VQVAEModel_33277406609677
// MI455X (gfx1250) — matrix operand order repaired and checked by reading, not compiled, not run on hardware
//
#include <hip/hip_runtime.h>

// ---------------------------------------------------------------- types
typedef __attribute__((ext_vector_type(8)))  float  v8f;
typedef __attribute__((ext_vector_type(8)))  __bf16 v8bf;
typedef __attribute__((ext_vector_type(16))) __bf16 v16bf;

// ---------------------------------------------------------------- sizes
#define BSZ     16384
#define IN_DIM  768
#define EDIM    500      // encoder dim
#define NPAD    512      // EDIM padded to 16
#define KCB     43       // codebook entries
#define VDIM    1024
#define CAT_K   1067     // VDIM + KCB
#define CAT_PAD 1088     // padded to 64

// d_out layout (float elements): x_tilde, z_e_x, z_q_x, logits, kl
#define OFF_XT  0
#define OFF_ZE  ((size_t)BSZ * IN_DIM)
#define OFF_ZQ  (OFF_ZE + (size_t)BSZ * EDIM)
#define OFF_LG  (OFF_ZQ + (size_t)BSZ * EDIM)
#define OFF_KL  (OFF_LG + (size_t)BSZ * KCB)

// ---------------------------------------------------------------- frag loads
// A-frag (16x32 bf16, MxK): lane li=lane&15 -> M, lh=lane>>4.
// elements 0..7  = K[8*lh .. 8*lh+7], elements 8..15 = K[16+8*lh .. 16+8*lh+7]
__device__ __forceinline__ v16bf frag_a(const __bf16* p, int lh) {
  v8bf lo = *(const v8bf*)(p + 8 * lh);
  v8bf hi = *(const v8bf*)(p + 16 + 8 * lh);
  return __builtin_shufflevector(lo, hi, 0,1,2,3,4,5,6,7,8,9,10,11,12,13,14,15);
}
// B-frag (32x16 bf16, KxN) sourced from row-major W (N,K): lane li -> N,
// elements 0..7  = K[8*lh .. 8*lh+7], elements 8..15 = K[16+8*lh .. 16+8*lh+7]
__device__ __forceinline__ v16bf frag_b(const __bf16* p, int lh) {
  v8bf lo = *(const v8bf*)(p + 8 * lh);
  v8bf hi = *(const v8bf*)(p + 16 + 8 * lh);
  return __builtin_shufflevector(lo, hi, 0,1,2,3,4,5,6,7,8,9,10,11,12,13,14,15);
}

__device__ __forceinline__ v8f vzero8() {
  v8f z = {0.f,0.f,0.f,0.f,0.f,0.f,0.f,0.f};
  return z;
}

// ---------------------------------------------------------------- weight cvt
__global__ __launch_bounds__(256) void cvt_pad(const float* __restrict__ src,
                                               __bf16* __restrict__ dst,
                                               int R, int C, int Rp, int Cp) {
  int idx = blockIdx.x * 256 + threadIdx.x;
  if (idx >= Rp * Cp) return;
  int r = idx / Cp, c = idx - r * Cp;
  float v = (r < R && c < C) ? src[(size_t)r * C + c] : 0.f;
  dst[idx] = (__bf16)v;
}

// ---------------------------------------------------------------- main GEMM
// C(MxN) = A(MxK) * Bw(N,K)^T + bias.  Block tile 128x256, BK=64, 8 waves in
// a 2x4 grid, wave tile 64x64 (4x4 WMMA 16x16x32 bf16 tiles, f32 accum):
// 16 ds_load_b128 feed 16 WMMAs per 32-K substep (1.0 loads/WMMA).
// EPI 0: store f32 (guarded col < nstore).  EPI 1: relu + store bf16.
template<int EPI, bool AF32>
__global__ __launch_bounds__(256)
void gemm_wmma(const void* __restrict__ Ap, int lda,
               const __bf16* __restrict__ Bw, int ldb,
               const float* __restrict__ bias,
               float* __restrict__ outF, __bf16* __restrict__ outB,
               int ldc, int nstore, int K) {
  constexpr int LDS = 72;                 // 64 + 8 pad (conflict-free frags)
  __shared__ __bf16 As[128 * LDS];        // 18.4 KB
  __shared__ __bf16 Bs[256 * LDS];        // 36.9 KB

  const int tid  = threadIdx.x;
  const int lane = tid & 31, wave = tid >> 5;
  const int li = lane & 15, lh = lane >> 4;
  const int wrow = wave >> 2, wcol = wave & 3;
  const int rowBase = blockIdx.x * 128;
  const int colBase = blockIdx.y * 256;
  const int lr = tid >> 1;                // A stage: 0..127 row
  const int lc = (tid & 1) * 32;          // A stage: 0/32 col segment

  v8f acc[4][4];
#pragma unroll
  for (int a = 0; a < 4; ++a)
#pragma unroll
    for (int b = 0; b < 4; ++b) acc[a][b] = vzero8();

  for (int k0 = 0; k0 < K; k0 += 64) {
    // ---- stage A (global -> bf16 LDS, f32 converted on the fly)
    if (AF32) {
      const float* Af = (const float*)Ap;
      const float4* s = (const float4*)(Af + (size_t)(rowBase + lr) * lda + k0 + lc);
#pragma unroll
      for (int j = 0; j < 4; ++j) {
        float4 f0 = s[2 * j], f1 = s[2 * j + 1];
        v8bf b = { (__bf16)f0.x, (__bf16)f0.y, (__bf16)f0.z, (__bf16)f0.w,
                   (__bf16)f1.x, (__bf16)f1.y, (__bf16)f1.z, (__bf16)f1.w };
        *(v8bf*)&As[lr * LDS + lc + 8 * j] = b;
      }
    } else {
      const __bf16* Ab = (const __bf16*)Ap;
      const uint4* s = (const uint4*)(Ab + (size_t)(rowBase + lr) * lda + k0 + lc);
#pragma unroll
      for (int j = 0; j < 4; ++j)
        *(uint4*)&As[lr * LDS + lc + 8 * j] = s[j];
    }
    // ---- stage B: one full 64-col row per thread (256 rows)
    {
      const uint4* s = (const uint4*)(Bw + (size_t)(colBase + tid) * ldb + k0);
#pragma unroll
      for (int j = 0; j < 8; ++j)
        *(uint4*)&Bs[tid * LDS + 8 * j] = s[j];
    }
    __syncthreads();
#pragma unroll
    for (int ks = 0; ks < 64; ks += 32) {
      v16bf af[4], bf[4];
#pragma unroll
      for (int tm = 0; tm < 4; ++tm)
        af[tm] = frag_a(&As[(wrow * 64 + tm * 16 + li) * LDS + ks], lh);
#pragma unroll
      for (int tn = 0; tn < 4; ++tn)
        bf[tn] = frag_b(&Bs[(wcol * 64 + tn * 16 + li) * LDS + ks], lh);
#pragma unroll
      for (int tm = 0; tm < 4; ++tm)
#pragma unroll
        for (int tn = 0; tn < 4; ++tn)
          acc[tm][tn] = __builtin_amdgcn_wmma_f32_16x16x32_bf16(
              false, af[tm], false, bf[tn], (short)0, acc[tm][tn], false, false);
    }
    __syncthreads();
  }

  // ---- epilogue: D layout: row = base + 8*lh + v, col = base + li
#pragma unroll
  for (int tm = 0; tm < 4; ++tm) {
#pragma unroll
    for (int tn = 0; tn < 4; ++tn) {
      const int col = colBase + wcol * 64 + tn * 16 + li;
      const float bv = (col < nstore) ? bias[col] : 0.f;
#pragma unroll
      for (int v = 0; v < 8; ++v) {
        const int row = rowBase + wrow * 64 + tm * 16 + lh * 8 + v;
        float val = acc[tm][tn][v] + bv;
        if (EPI == 0) {
          if (col < nstore) outF[(size_t)row * ldc + col] = val;
        } else {
          outB[(size_t)row * ldc + col] = (__bf16)fmaxf(val, 0.f);
        }
      }
    }
  }
}

// ---------------------------------------------------------------- VAE GEMM
// Computes matching mu (cols j..j+63) and logvar (cols 1024+j..) tiles so the
// epilogue can fuse KL + reparameterized sampling. Block 128x64 per half,
// 8 waves in 4x2 grid, wave tile 32x32 per half (2x2 tiles x 2 halves).
__global__ __launch_bounds__(256)
void gemm_vae(const __bf16* __restrict__ h, const __bf16* __restrict__ W2b,
              const float* __restrict__ b2, const float* __restrict__ eps,
              __bf16* __restrict__ A4, float* __restrict__ klp) {
  constexpr int LDS = 72;
  __shared__ __bf16 As[128 * LDS];
  __shared__ __bf16 Bs0[64 * LDS];
  __shared__ __bf16 Bs1[64 * LDS];

  const int tid  = threadIdx.x;
  const int lane = tid & 31, wave = tid >> 5;
  const int li = lane & 15, lh = lane >> 4;
  const int wrow = wave >> 1, wcol = wave & 1;
  const int rowBase = blockIdx.x * 128;
  const int colBase = blockIdx.y * 64;
  const int lr = tid >> 1;                // A stage: 0..127
  const int lc = (tid & 1) * 32;
  const int nb = tid >> 2;                // B stage: 0..63
  const int cseg = (tid & 3) * 16;

  v8f acc0[2][2], acc1[2][2];
#pragma unroll
  for (int a = 0; a < 2; ++a)
#pragma unroll
    for (int b = 0; b < 2; ++b) { acc0[a][b] = vzero8(); acc1[a][b] = vzero8(); }

  for (int k0 = 0; k0 < VDIM; k0 += 64) {
    const uint4* sa = (const uint4*)(h + (size_t)(rowBase + lr) * VDIM + k0 + lc);
#pragma unroll
    for (int j = 0; j < 4; ++j)
      *(uint4*)&As[lr * LDS + lc + 8 * j] = sa[j];
    const uint4* s0 = (const uint4*)(W2b + (size_t)(colBase + nb) * VDIM + k0 + cseg);
    *(uint4*)&Bs0[nb * LDS + cseg]     = s0[0];
    *(uint4*)&Bs0[nb * LDS + cseg + 8] = s0[1];
    const uint4* s1 = (const uint4*)(W2b + (size_t)(VDIM + colBase + nb) * VDIM + k0 + cseg);
    *(uint4*)&Bs1[nb * LDS + cseg]     = s1[0];
    *(uint4*)&Bs1[nb * LDS + cseg + 8] = s1[1];
    __syncthreads();
#pragma unroll
    for (int ks = 0; ks < 64; ks += 32) {
      v16bf af[2], b0[2], b1[2];
#pragma unroll
      for (int tm = 0; tm < 2; ++tm)
        af[tm] = frag_a(&As[(wrow * 32 + tm * 16 + li) * LDS + ks], lh);
#pragma unroll
      for (int tn = 0; tn < 2; ++tn) {
        b0[tn] = frag_b(&Bs0[(wcol * 32 + tn * 16 + li) * LDS + ks], lh);
        b1[tn] = frag_b(&Bs1[(wcol * 32 + tn * 16 + li) * LDS + ks], lh);
      }
#pragma unroll
      for (int tm = 0; tm < 2; ++tm)
#pragma unroll
        for (int tn = 0; tn < 2; ++tn) {
          acc0[tm][tn] = __builtin_amdgcn_wmma_f32_16x16x32_bf16(
              false, af[tm], false, b0[tn], (short)0, acc0[tm][tn], false, false);
          acc1[tm][tn] = __builtin_amdgcn_wmma_f32_16x16x32_bf16(
              false, af[tm], false, b1[tn], (short)0, acc1[tm][tn], false, false);
        }
    }
    __syncthreads();
  }

  float kl = 0.f;
#pragma unroll
  for (int tm = 0; tm < 2; ++tm) {
#pragma unroll
    for (int tn = 0; tn < 2; ++tn) {
      const int col = colBase + wcol * 32 + tn * 16 + li;
      const float bmu = b2[col], blv = b2[col + VDIM];
#pragma unroll
      for (int v = 0; v < 8; ++v) {
        const int row = rowBase + wrow * 32 + tm * 16 + lh * 8 + v;
        float mu = acc0[tm][tn][v] + bmu;
        float lv = acc1[tm][tn][v] + blv;
        float elv = expf(lv);
        kl += elv + mu * mu - 1.f - lv;
        float sp = fmaf(sqrtf(elv), eps[(size_t)row * VDIM + col], mu);
        A4[(size_t)row * CAT_PAD + col] = (__bf16)sp;
      }
    }
  }
#pragma unroll
  for (int off = 16; off; off >>= 1) kl += __shfl_xor(kl, off, 32);
  if (lane == 0) atomicAdd(klp, kl * (0.5f / (float)BSZ));
}

// ---------------------------------------------------------------- codebook
// One wave per row: logits = z_e . codebook^T (coalesced), butterfly reduce,
// streaming softmax+argmax, writes logits/z_q_x and bf16 y into concat buffer.
__global__ __launch_bounds__(256)
void cb_kernel(const float* __restrict__ z_e, const float* __restrict__ codebook,
               const int* __restrict__ labels, const int* __restrict__ known,
               float* __restrict__ logits, float* __restrict__ zq,
               __bf16* __restrict__ A4) {
  __shared__ float lbuf[8][48];
  const int tid = threadIdx.x, wave = tid >> 5, lane = tid & 31;
  const int row = blockIdx.x * 8 + wave;
  const float* z = z_e + (size_t)row * EDIM;

  float part[KCB];
#pragma unroll
  for (int c = 0; c < KCB; ++c) part[c] = 0.f;

  for (int d0 = 0; d0 < EDIM; d0 += 32) {
    const int d = d0 + lane;
    const bool ok = d < EDIM;
    const float zc = ok ? z[d] : 0.f;
#pragma unroll
    for (int c = 0; c < KCB; ++c) {
      const float cv = ok ? codebook[(size_t)c * EDIM + d] : 0.f;
      part[c] = fmaf(zc, cv, part[c]);
    }
  }

  float m = -__builtin_inff(), S = 0.f;
  int amax = 0;
#pragma unroll
  for (int c = 0; c < KCB; ++c) {
    float v = part[c];
#pragma unroll
    for (int off = 16; off; off >>= 1) v += __shfl_xor(v, off, 32);
    if (lane == 0) {
      logits[(size_t)row * KCB + c] = v;
      lbuf[wave][c] = v;
    }
    if (v > m) { S = S * expf(m - v) + 1.f; m = v; amax = c; }
    else       { S += expf(v - m); }
  }
  __syncthreads();

  const int lbl = labels[row];
  const int km  = known[row];
#pragma unroll
  for (int c0 = 0; c0 < 64; c0 += 32) {
    const int c = c0 + lane;
    float yv = 0.f;
    if (c < KCB)
      yv = km ? ((c == lbl) ? 1.f : 0.f) : expf(lbuf[wave][c] - m) / S;
    A4[(size_t)row * CAT_PAD + VDIM + c] = (__bf16)yv;   // cols >= 1067 zeroed
  }
  const float* crow = codebook + (size_t)amax * EDIM;
  for (int d = lane; d < EDIM; d += 32)
    zq[(size_t)row * EDIM + d] = crow[d];
}

// ---------------------------------------------------------------- launch
extern "C" void kernel_launch(void* const* d_in, const int* in_sizes, int n_in,
                              void* d_out, int out_size, void* d_ws, size_t ws_size,
                              hipStream_t stream) {
  const float* x        = (const float*)d_in[0];
  const int*   known    = (const int*)  d_in[1];
  const int*   labels   = (const int*)  d_in[2];
  const float* eps      = (const float*)d_in[3];
  const float* W_enc    = (const float*)d_in[4];
  const float* b_enc    = (const float*)d_in[5];
  const float* codebook = (const float*)d_in[6];
  const float* W1       = (const float*)d_in[7];
  const float* b1       = (const float*)d_in[8];
  const float* W2       = (const float*)d_in[9];
  const float* b2       = (const float*)d_in[10];
  const float* Wd       = (const float*)d_in[11];
  const float* bd       = (const float*)d_in[12];
  float* out = (float*)d_out;
  char*  ws  = (char*)d_ws;

  // workspace layout (bytes)
  __bf16* WencB = (__bf16*)(ws);                                  // 512*768
  __bf16* W1B   = (__bf16*)(ws + 786432);                         // 1024*768
  __bf16* W2B   = (__bf16*)(ws + 786432 + 1572864);               // 2048*1024
  __bf16* WdB   = (__bf16*)(ws + 786432 + 1572864 + 4194304);     // 768*1088
  __bf16* hB    = (__bf16*)(ws + 786432 + 1572864 + 4194304 + 1671168);
  __bf16* A4B   = (__bf16*)(ws + 786432 + 1572864 + 4194304 + 1671168 + 33554432);

  // 1) weights -> bf16 (with zero padding)
  cvt_pad<<<(NPAD * IN_DIM + 255) / 256, 256, 0, stream>>>(W_enc, WencB, EDIM, IN_DIM, NPAD, IN_DIM);
  cvt_pad<<<(VDIM * IN_DIM + 255) / 256, 256, 0, stream>>>(W1, W1B, VDIM, IN_DIM, VDIM, IN_DIM);
  cvt_pad<<<(2 * VDIM * VDIM + 255) / 256, 256, 0, stream>>>(W2, W2B, 2 * VDIM, VDIM, 2 * VDIM, VDIM);
  cvt_pad<<<(IN_DIM * CAT_PAD + 255) / 256, 256, 0, stream>>>(Wd, WdB, IN_DIM, CAT_K, IN_DIM, CAT_PAD);

  // 2) zero kl accumulator
  hipMemsetAsync(out + OFF_KL, 0, sizeof(float), stream);

  // 3) z_e_x = x @ W_enc^T + b_enc   (f32 out, N padded 512, store 500)
  gemm_wmma<0, true><<<dim3(BSZ / 128, NPAD / 256), 256, 0, stream>>>(
      x, IN_DIM, WencB, IN_DIM, b_enc, out + OFF_ZE, nullptr, EDIM, EDIM, IN_DIM);

  // 4) logits / softmax / argmax / z_q_x / y -> concat cols [1024,1088)
  cb_kernel<<<BSZ / 8, 256, 0, stream>>>(out + OFF_ZE, codebook, labels, known,
                                         out + OFF_LG, out + OFF_ZQ, A4B);

  // 5) h = relu(x @ W1^T + b1) -> bf16
  gemm_wmma<1, true><<<dim3(BSZ / 128, VDIM / 256), 256, 0, stream>>>(
      x, IN_DIM, W1B, IN_DIM, b1, nullptr, hB, VDIM, VDIM, IN_DIM);

  // 6) mu/logvar + KL + sampled_z -> concat cols [0,1024)
  gemm_vae<<<dim3(BSZ / 128, VDIM / 64), 256, 0, stream>>>(
      hB, W2B, b2, eps, A4B, out + OFF_KL);

  // 7) x_tilde = concat @ Wd^T + bd   (K padded 1088)
  gemm_wmma<0, false><<<dim3(BSZ / 128, IN_DIM / 256), 256, 0, stream>>>(
      A4B, CAT_PAD, WdB, CAT_PAD, bd, out + OFF_XT, nullptr, IN_DIM, IN_DIM, CAT_PAD);
}
